// mixprop_6725918785817
// MI455X (gfx1250) — matrix operand order repaired and checked by reading, not compiled, not run on hardware
//
#include <hip/hip_runtime.h>

// mixprop for MI455X (gfx1250, wave32, WMMA).
// Key optimization: beta*nconv(h,a1)+gamma*nconv(h,a2) == nconv(h, beta*a1+gamma*a2),
// so we precompute one combined bf16 adjacency and do ONE 512x512 x 512x86016 GEMM
// per iteration using v_wmma_f32_16x16x32_bf16 (bf16 in, f32 accumulate).
//
// Workspace layout (requires ~354 MB):
//   [0, 512KB)          : combined adjacency A, bf16, row-major [512][512]
//   [1MB, 1MB+176MB)    : h ping buffer, f32, [N,C,V,L]
//   [.., +176MB)        : h pong buffer, f32, [N,C,V,L]

#define NN   16
#define CC   32
#define VV   512
#define LL   168
#define LP   176            // L padded to multiple of 16
#define KT   32             // WMMA K per step
#define BM   64             // rows (v) per workgroup in nconv
#define BSTR 40             // padded K-stride for B tiles in LDS (80B, 16B aligned)
#define VL   ((size_t)VV * LL)      // 86016
#define ALPHA_C 0.05f
#define BETA_C  0.05f
#define GAMMA_C 0.90f

typedef __bf16 bf16_t;
typedef bf16_t v16bf __attribute__((ext_vector_type(16)));
typedef float  v8f   __attribute__((ext_vector_type(8)));

union Frag { v16bf v; uint4 q[2]; };

// ---------------------------------------------------------------------------
// Kernel 1: combined normalized adjacency, cast to bf16.
// A[v][w] = BETA*(adj1[v][w]+I)/rowsum1(v) + GAMMA*(adj2[v][w]+I)/rowsum2(v)
// ---------------------------------------------------------------------------
__global__ __launch_bounds__(256)
void prep_adj_kernel(const float* __restrict__ adj1, const float* __restrict__ adj2,
                     bf16_t* __restrict__ Abf) {
    const int v = blockIdx.x;
    const int tid = threadIdx.x;
    __shared__ float r1[256];
    __shared__ float r2[256];
    float s1 = 0.0f, s2 = 0.0f;
    for (int w = tid; w < VV; w += 256) {
        s1 += adj1[v * VV + w];
        s2 += adj2[v * VV + w];
    }
    r1[tid] = s1; r2[tid] = s2;
    __syncthreads();
    for (int off = 128; off > 0; off >>= 1) {
        if (tid < off) { r1[tid] += r1[tid + off]; r2[tid] += r2[tid + off]; }
        __syncthreads();
    }
    const float inv1 = BETA_C  / (r1[0] + 1.0f);   // +1: identity added to diagonal
    const float inv2 = GAMMA_C / (r2[0] + 1.0f);
    for (int w = tid; w < VV; w += 256) {
        float e1 = adj1[v * VV + w] + (w == v ? 1.0f : 0.0f);
        float e2 = adj2[v * VV + w] + (w == v ? 1.0f : 0.0f);
        Abf[(size_t)v * VV + w] = (bf16_t)(e1 * inv1 + e2 * inv2);
    }
}

// ---------------------------------------------------------------------------
// Kernel 2: hnext = ALPHA*x + A @ hprev   (per (n,c) slab: C[512x168] = A[512x512] H[512x168])
// Workgroup: 4 waves, BM=64 rows x full L (padded 176). 11 WMMAs/wave/K-step.
// ---------------------------------------------------------------------------
__global__ __launch_bounds__(128)
void nconv_kernel(const bf16_t* __restrict__ Abf,
                  const float* __restrict__ hprev,
                  const float* __restrict__ x,
                  float* __restrict__ hnext) {
    __shared__ bf16_t a_lds[BM * KT];      // [m][k], row-major, k contiguous
    __shared__ bf16_t b_lds[LP * BSTR];    // [l][k], k contiguous, padded stride

    const int v0   = blockIdx.x * BM;
    const int nc   = blockIdx.y;
    const size_t slab = (size_t)nc * VL;
    const int tid  = threadIdx.x;
    const int lane = tid & 31;
    const int wave = tid >> 5;
    const int mlane = lane & 15;
    const int hi    = lane >> 4;           // 0: lanes 0-15, 1: lanes 16-31

    v8f acc[11];
#pragma unroll
    for (int i = 0; i < 11; ++i) acc[i] = (v8f){0.f,0.f,0.f,0.f,0.f,0.f,0.f,0.f};

    for (int k0 = 0; k0 < VV; k0 += KT) {
        // --- stage A tile: 64x32 bf16, 16B chunks (8 bf16) ---
        for (int ch = tid; ch < (BM * KT) / 8; ch += 128) {
            const int m  = ch >> 2;
            const int k8 = (ch & 3) * 8;
            *(uint4*)(&a_lds[m * KT + k8]) =
                *(const uint4*)(Abf + (size_t)(v0 + m) * VV + k0 + k8);
        }
        // --- stage B tile transposed: [l][k], zero-pad l in [168,176) ---
        for (int idx = tid; idx < KT * LP; idx += 128) {
            const int k = idx / LP;
            const int l = idx - k * LP;
            const float val = (l < LL) ? hprev[slab + (size_t)(k0 + k) * LL + l] : 0.0f;
            b_lds[l * BSTR + k] = (bf16_t)val;
        }
        __syncthreads();

        // A fragment: 16x32, ISA layout (lanes 0-15: K {0..7,16..23}; 16-31: K {8..15,24..31})
        Frag af;
        const int m = wave * 16 + mlane;
        af.q[0] = *(const uint4*)&a_lds[m * KT + hi * 8];
        af.q[1] = *(const uint4*)&a_lds[m * KT + 16 + hi * 8];

#pragma unroll
        for (int ni = 0; ni < 11; ++ni) {
            Frag bfg;
            const int col = ni * 16 + mlane;   // column = lane&15; lanes 16-31: K {8..15,24..31}
            bfg.q[0] = *(const uint4*)&b_lds[col * BSTR + hi * 8];
            bfg.q[1] = *(const uint4*)&b_lds[col * BSTR + 16 + hi * 8];
            acc[ni] = __builtin_amdgcn_wmma_f32_16x16x32_bf16(
                false, af.v, false, bfg.v, (short)0, acc[ni], false, false);
        }
        __syncthreads();
    }

    // epilogue: D layout: VGPR r -> M = r + hi*8, N = lane&15
#pragma unroll
    for (int ni = 0; ni < 11; ++ni) {
        const int l = ni * 16 + mlane;
        if (l < LL) {
            const int vbase = v0 + wave * 16 + hi * 8;
#pragma unroll
            for (int r = 0; r < 8; ++r) {
                const size_t idx = slab + (size_t)(vbase + r) * LL + l;
                hnext[idx] = ALPHA_C * x[idx] + acc[ni][r];
            }
        }
    }
}

// ---------------------------------------------------------------------------
// Kernel 3: 1x1 conv stage: y[n,o,pos] (+)= sum_c W[o, stage*32+c] * h[n,c,pos] (+ b at stage 0)
// Per workgroup: one n, 256 positions. M=32 (2 tiles), K=32 (one WMMA step).
// ---------------------------------------------------------------------------
__global__ __launch_bounds__(128)
void yconv_kernel(const float* __restrict__ h, const float* __restrict__ W,
                  const float* __restrict__ bias, float* __restrict__ y, int stage) {
    __shared__ bf16_t w_lds[CC * KT];        // [o][c]
    __shared__ bf16_t h_lds[256 * BSTR];     // [p][c], padded stride

    const int n  = blockIdx.y;
    const int p0 = blockIdx.x * 256;         // 86016 = 256*336, no tail
    const int tid  = threadIdx.x;
    const int lane = tid & 31;
    const int wave = tid >> 5;
    const int mlane = lane & 15;
    const int hi    = lane >> 4;

    for (int idx = tid; idx < CC * KT; idx += 128) {
        const int o = idx >> 5, c = idx & 31;
        w_lds[o * KT + c] = (bf16_t)W[o * 128 + stage * 32 + c];
    }
    for (int idx = tid; idx < 256 * KT; idx += 128) {
        const int p = idx & 255, c = idx >> 8;
        h_lds[p * BSTR + c] = (bf16_t)h[((size_t)(n * CC + c)) * VL + p0 + p];
    }
    __syncthreads();

#pragma unroll
    for (int mi = 0; mi < 2; ++mi) {
        Frag af;
        const int o = mi * 16 + mlane;
        af.q[0] = *(const uint4*)&w_lds[o * KT + hi * 8];
        af.q[1] = *(const uint4*)&w_lds[o * KT + 16 + hi * 8];
#pragma unroll
        for (int j = 0; j < 4; ++j) {
            Frag bfg;
            const int col = (wave * 4 + j) * 16 + mlane;
            bfg.q[0] = *(const uint4*)&h_lds[col * BSTR + hi * 8];
            bfg.q[1] = *(const uint4*)&h_lds[col * BSTR + 16 + hi * 8];
            v8f acc = (v8f){0.f,0.f,0.f,0.f,0.f,0.f,0.f,0.f};
            acc = __builtin_amdgcn_wmma_f32_16x16x32_bf16(
                false, af.v, false, bfg.v, (short)0, acc, false, false);
            const int obase = mi * 16 + hi * 8;
            const size_t pg = (size_t)p0 + (wave * 4 + j) * 16 + mlane;
#pragma unroll
            for (int r = 0; r < 8; ++r) {
                const size_t idx = ((size_t)(n * CC + obase + r)) * VL + pg;
                if (stage == 0) y[idx] = bias[obase + r] + acc[r];
                else            y[idx] += acc[r];
            }
        }
    }
}

// ---------------------------------------------------------------------------
extern "C" void kernel_launch(void* const* d_in, const int* in_sizes, int n_in,
                              void* d_out, int out_size, void* d_ws, size_t ws_size,
                              hipStream_t stream) {
    const float* x    = (const float*)d_in[0];
    const float* adj1 = (const float*)d_in[1];
    const float* adj2 = (const float*)d_in[2];
    const float* W    = (const float*)d_in[3];
    const float* b    = (const float*)d_in[4];
    float* y = (float*)d_out;

    char* ws = (char*)d_ws;
    bf16_t* Abf = (bf16_t*)ws;                                   // 512*512*2 = 512 KB
    const size_t HBUF = (size_t)NN * CC * VV * LL;               // 44,040,192 floats
    float* buf0 = (float*)(ws + (1u << 20));
    float* buf1 = buf0 + HBUF;

    const dim3 gGemm(VV / BM, NN * CC);          // (8, 512)
    const dim3 gY(VL / 256, NN);                 // (336, 16)

    prep_adj_kernel<<<VV, 256, 0, stream>>>(adj1, adj2, Abf);

    // stage 0: y = b + W0 @ x
    yconv_kernel<<<gY, 128, 0, stream>>>(x, W, b, y, 0);
    // h1 = ALPHA*x + A @ x
    nconv_kernel<<<gGemm, 128, 0, stream>>>(Abf, x, x, buf0);
    yconv_kernel<<<gY, 128, 0, stream>>>(buf0, W, b, y, 1);
    // h2 = ALPHA*x + A @ h1
    nconv_kernel<<<gGemm, 128, 0, stream>>>(Abf, buf0, x, buf1);
    yconv_kernel<<<gY, 128, 0, stream>>>(buf1, W, b, y, 2);
    // h3 = ALPHA*x + A @ h2   (buf0 / h1 already consumed)
    nconv_kernel<<<gGemm, 128, 0, stream>>>(Abf, buf1, x, buf0);
    yconv_kernel<<<gY, 128, 0, stream>>>(buf0, W, b, y, 3);
}
